// ContextPooling_7172595385030
// MI455X (gfx1250) — matrix operand order repaired and checked by reading, not compiled, not run on hardware
//
#include <hip/hip_runtime.h>
#include <hip/hip_bf16.h>

typedef __attribute__((ext_vector_type(16))) _Float16 v16h;
typedef __attribute__((ext_vector_type(8)))  float    v8f;

#define NB   16
#define LSEQ 1024
#define DIN  768
#define D2   1536
#define LPAD 1032   // 1 left pad + 1024 data + 7 right pad -> rows stay 16B aligned
#define LDT  40     // padded LDS stride (halves): 80B rows, conflict-friendly

// ---------------------------------------------------------------------------
// WMMA fragment loaders (wave32 layouts per CDNA5 ISA 7.12.2), vectorized:
// each lane issues 2x ds_load_b128 per fragment.
// ---------------------------------------------------------------------------
__device__ __forceinline__ v16h load_a_frag(const _Float16* p, int lda) {
    const int lane = threadIdx.x & 31;
    const int m  = lane & 15;
    const int kb = (lane >> 4) << 3;      // 0 or 8
    const _Float16* r = p + m * lda + kb;
    union { v16h v; uint4 q[2]; } u;
    u.q[0] = *(const uint4*)r;            // halves 0..7  : K kb..kb+7
    u.q[1] = *(const uint4*)(r + 16);     // halves 8..15 : K kb+16..kb+23
    return u.v;
}

// B staged TRANSPOSED in LDS: Bt[col][k], row stride LDT halves.
// lane<16: N=lane, K 0..7,16..23 ; lane>=16: N=lane-16, K 8..15,24..31.
__device__ __forceinline__ v16h load_bt_frag(const _Float16* bt, int col0) {
    const int lane = threadIdx.x & 31;
    const int n  = lane & 15;
    const int kb = (lane >> 4) << 3;      // 0 or 8
    const _Float16* c = bt + (size_t)(col0 + n) * LDT + kb;
    union { v16h v; uint4 q[2]; } u;
    u.q[0] = *(const uint4*)c;
    u.q[1] = *(const uint4*)(c + 16);
    return u.v;
}

__device__ __forceinline__ float block_reduce_sum(float v, float* red) {
    const int tid = threadIdx.x;
    red[tid] = v; __syncthreads();
    for (int s = 128; s > 0; s >>= 1) { if (tid < s) red[tid] += red[tid + s]; __syncthreads(); }
    float r = red[0]; __syncthreads();
    return r;
}
__device__ __forceinline__ float block_reduce_max(float v, float* red) {
    const int tid = threadIdx.x;
    red[tid] = v; __syncthreads();
    for (int s = 128; s > 0; s >>= 1) { if (tid < s) red[tid] = fmaxf(red[tid], red[tid + s]); __syncthreads(); }
    float r = red[0]; __syncthreads();
    return r;
}

// ---------------------------------------------------------------------------
// Init: zero LN stats + pad columns of the three padded activation buffers
// ---------------------------------------------------------------------------
__global__ void init_kernel(float* stats, _Float16* xp0, _Float16* xp1, _Float16* xp2) {
    size_t idx = (size_t)blockIdx.x * blockDim.x + threadIdx.x;
    if (idx < 64) stats[idx] = 0.0f;
    const size_t n0 = (size_t)NB * DIN * 8, n1 = (size_t)NB * D2 * 8;
    const size_t total = n0 + 2 * n1;
    const size_t stride = (size_t)gridDim.x * blockDim.x;
    for (size_t t = idx; t < total; t += stride) {
        _Float16* buf; size_t u;
        if (t < n0)           { buf = xp0; u = t; }
        else if (t < n0 + n1) { buf = xp1; u = t - n0; }
        else                  { buf = xp2; u = t - n0 - n1; }
        size_t r = u >> 3, p = u & 7;
        size_t col = p ? (1024 + p) : 0;          // col 0 and cols 1025..1031
        buf[r * LPAD + col] = (_Float16)0.0f;
    }
}

// conv_w [O][I][3] f32  ->  Wp [3][O][I] f16
__global__ void pack_w_kernel(const float* __restrict__ w, _Float16* __restrict__ o, int O, int I) {
    const size_t OI = (size_t)O * I, n = 3 * OI;
    const size_t stride = (size_t)gridDim.x * blockDim.x;
    for (size_t idx = (size_t)blockIdx.x * blockDim.x + threadIdx.x; idx < n; idx += stride) {
        size_t k = idx / OI, rem = idx - k * OI;
        size_t oc = rem / I, ic = rem - oc * I;
        o[idx] = (_Float16)w[(oc * I + ic) * 3 + k];
    }
}

// feats [B][L][D] f32 -> Xpad0 [B][D][LPAD] f16 (interior) + feats16 [B][L][D] f16
__global__ void pack_feats_kernel(const float* __restrict__ f, _Float16* __restrict__ xp,
                                  _Float16* __restrict__ f16) {
    const size_t n = (size_t)NB * LSEQ * DIN;
    const size_t stride = (size_t)gridDim.x * blockDim.x;
    for (size_t idx = (size_t)blockIdx.x * blockDim.x + threadIdx.x; idx < n; idx += stride) {
        float v = f[idx];
        f16[idx] = (_Float16)v;
        size_t d = idx % DIN, bl = idx / DIN;
        size_t l = bl % LSEQ, b = bl / LSEQ;
        xp[((b * DIN) + d) * LPAD + 1 + l] = (_Float16)v;
    }
}

// ---------------------------------------------------------------------------
// Conv-as-GEMM: Y[b,o,l] = bias[o] + sum_k sum_c Wk[o,c] * Xpad[b,c,l+k]
// block = 128 thr (4 waves), tile 64(O) x 64(L), K-chunks of 32 channels.
// B tile staged transposed in LDS so fragment reads are ds_load_b128.
// Next-chunk tiles prefetched (global_prefetch_b8) while current computes.
// ---------------------------------------------------------------------------
__global__ __launch_bounds__(128) void conv_gemm_kernel(
    const _Float16* __restrict__ Wp,   // [3][O][Cin]
    const _Float16* __restrict__ Xp,   // [B][Cin][LPAD]
    const float*    __restrict__ bias, // [O]
    _Float16*       __restrict__ Yp,   // [B][O][LPAD] (interior)
    int O, int Cin)
{
    __shared__ __align__(16) _Float16 As[3][64][LDT];  // row-major, K contiguous
    __shared__ __align__(16) _Float16 Bt[72][LDT];     // TRANSPOSED: [col][k]
    const int l0 = blockIdx.x * 64;
    const int o0 = blockIdx.y * 64;
    const int b  = blockIdx.z;
    const int tid  = threadIdx.x;
    const int wave = tid >> 5;
    const int lane = tid & 31;
    const int mrow = (lane >> 4) << 3;

    v8f acc[4];
#pragma unroll
    for (int r = 0; r < 8; ++r) {
        float bv = bias[o0 + wave * 16 + mrow + r];
#pragma unroll
        for (int j = 0; j < 4; ++j) acc[j][r] = bv;
    }

    const int nK = Cin >> 5;
    for (int kc = 0; kc < nK; ++kc) {
        // stage B transposed: 32 channel-rows x 72 padded cols (covers the 3 shifts)
        {
            const _Float16* src = Xp + ((size_t)b * Cin + ((size_t)kc << 5)) * LPAD + l0;
            for (int idx = tid; idx < 32 * 9; idx += 128) {
                int r = idx / 9, cc = idx - r * 9;
                uint4 q = *(const uint4*)(src + (size_t)r * LPAD + cc * 8);
                const _Float16* h = (const _Float16*)&q;
#pragma unroll
                for (int i = 0; i < 8; ++i) Bt[cc * 8 + i][r] = h[i];
            }
        }
        // stage A row-major: 3 shift-weight tiles of 64 x 32
        for (int idx = tid; idx < 3 * 64 * 4; idx += 128) {
            int k = idx >> 8, rem = idx & 255, r = rem >> 2, cc = rem & 3;
            const _Float16* s8 = Wp + ((size_t)k * O + o0 + r) * Cin + ((size_t)kc << 5) + cc * 8;
            *(uint4*)&As[k][r][cc * 8] = *(const uint4*)s8;
        }
        // prefetch next K-chunk into cache while this one computes
        if (kc + 1 < nK) {
            if (tid < 32) {
                __builtin_prefetch(Xp + ((size_t)b * Cin + (((size_t)kc + 1) << 5) + tid) * LPAD + l0, 0, 3);
            } else {
                int r = tid - 32;              // 0..95 -> rows r and r+96 of 3*64
                int k0 = r >> 6,        r0 = r & 63;
                int k1 = (r + 96) >> 6, r1 = (r + 96) & 63;
                __builtin_prefetch(Wp + ((size_t)k0 * O + o0 + r0) * Cin + (((size_t)kc + 1) << 5), 0, 3);
                __builtin_prefetch(Wp + ((size_t)k1 * O + o0 + r1) * Cin + (((size_t)kc + 1) << 5), 0, 3);
            }
        }
        __syncthreads();
#pragma unroll
        for (int k = 0; k < 3; ++k) {
            v16h a = load_a_frag(&As[k][wave * 16][0], LDT);
#pragma unroll
            for (int j = 0; j < 4; ++j) {
                v16h bf = load_bt_frag(&Bt[0][0], j * 16 + k);
                acc[j] = __builtin_amdgcn_wmma_f32_16x16x32_f16(
                    false, a, false, bf, (short)0, acc[j], false, false);
            }
        }
        __syncthreads();
    }

    _Float16* dst = Yp + ((size_t)b * O + o0 + wave * 16 + mrow) * LPAD + 1 + l0 + (lane & 15);
#pragma unroll
    for (int j = 0; j < 4; ++j)
#pragma unroll
        for (int r = 0; r < 8; ++r)
            dst[(size_t)r * LPAD + j * 16] = (_Float16)acc[j][r];
}

// ---------------------------------------------------------------------------
// LayerNorm over (C,L) jointly per batch: stats pass + in-place apply
// ---------------------------------------------------------------------------
__global__ __launch_bounds__(256) void ln_stats_kernel(const _Float16* __restrict__ X,
                                                       float* stats, int C) {
    __shared__ float red[256];
    const int b = blockIdx.y;
    const size_t total = (size_t)C * LSEQ;
    float s = 0.f, q = 0.f;
    for (size_t idx = (size_t)blockIdx.x * 256 + threadIdx.x; idx < total;
         idx += (size_t)gridDim.x * 256) {
        size_t c = idx >> 10, l = idx & 1023;
        float v = (float)X[((size_t)b * C + c) * LPAD + 1 + l];
        s += v; q += v * v;
    }
    float ts = block_reduce_sum(s, red);
    float tq = block_reduce_sum(q, red);
    if (threadIdx.x == 0) {
        atomicAdd(&stats[b * 2 + 0], ts);
        atomicAdd(&stats[b * 2 + 1], tq);
    }
}

__global__ __launch_bounds__(256) void ln_apply_kernel(_Float16* __restrict__ X,
                                                       const float* __restrict__ stats,
                                                       const float* __restrict__ g,
                                                       const float* __restrict__ be, int C) {
    const int b = blockIdx.y;
    const size_t total = (size_t)C * LSEQ;
    const float invN = 1.0f / (float)total;
    float mean = stats[b * 2 + 0] * invN;
    float var  = stats[b * 2 + 1] * invN - mean * mean;
    float rstd = rsqrtf(var + 1e-5f);
    for (size_t idx = (size_t)blockIdx.x * 256 + threadIdx.x; idx < total;
         idx += (size_t)gridDim.x * 256) {
        size_t c = idx >> 10, l = idx & 1023;
        size_t a = ((size_t)b * C + c) * LPAD + 1 + l;
        float v = (float)X[a];
        X[a] = (_Float16)((v - mean) * rstd * g[idx] + be[idx]);
    }
}

// ---------------------------------------------------------------------------
// Head: conv3 (2 out channels) + LN over (2,L) + per-channel softmax -> s, w
// one block per batch
// ---------------------------------------------------------------------------
__global__ __launch_bounds__(256) void head_kernel(
    const _Float16* __restrict__ X,  // Xpad2 [B][D2][LPAD]
    const float* __restrict__ w3, const float* __restrict__ b3,
    const float* __restrict__ g3, const float* __restrict__ be3,
    float* __restrict__ sout, float* __restrict__ wout)
{
    __shared__ float w3s[2 * D2 * 3];
    __shared__ float t[2][LSEQ];
    __shared__ float red[256];
    const int b = blockIdx.x, tid = threadIdx.x;
    for (int i = tid; i < 2 * D2 * 3; i += 256) w3s[i] = w3[i];
    __syncthreads();

    const _Float16* xb = X + (size_t)b * D2 * LPAD;
    for (int l = tid; l < LSEQ; l += 256) {
        float a0 = b3[0], a1 = b3[1];
        for (int c = 0; c < D2; ++c) {
            const _Float16* xr = xb + (size_t)c * LPAD + l;   // padded cols l..l+2
            float x0 = (float)xr[0], x1 = (float)xr[1], x2 = (float)xr[2];
            const float* wa = &w3s[c * 3];
            const float* wb = &w3s[(D2 + c) * 3];
            a0 = fmaf(wa[0], x0, fmaf(wa[1], x1, fmaf(wa[2], x2, a0)));
            a1 = fmaf(wb[0], x0, fmaf(wb[1], x1, fmaf(wb[2], x2, a1)));
        }
        t[0][l] = a0; t[1][l] = a1;
    }
    __syncthreads();

    // LayerNorm jointly over 2*1024 values
    float ls = 0.f, lq = 0.f;
    for (int idx = tid; idx < 2 * LSEQ; idx += 256) {
        float v = t[idx >> 10][idx & 1023]; ls += v; lq += v * v;
    }
    float sum  = block_reduce_sum(ls, red);
    float sumq = block_reduce_sum(lq, red);
    float mean = sum / 2048.f;
    float var  = sumq / 2048.f - mean * mean;
    float rstd = rsqrtf(var + 1e-5f);
    for (int idx = tid; idx < 2 * LSEQ; idx += 256) {
        int ch = idx >> 10, l = idx & 1023;
        t[ch][l] = (t[ch][l] - mean) * rstd * g3[idx] + be3[idx];
    }
    __syncthreads();

    // softmax per channel
    for (int ch = 0; ch < 2; ++ch) {
        float m = -1e30f;
        for (int l = tid; l < LSEQ; l += 256) m = fmaxf(m, t[ch][l]);
        m = block_reduce_max(m, red);
        float e = 0.f;
        for (int l = tid; l < LSEQ; l += 256) e += __expf(t[ch][l] - m);
        e = block_reduce_sum(e, red);
        float inv = 1.0f / e;
        float* o = ch ? wout : sout;
        for (int l = tid; l < LSEQ; l += 256) o[b * LSEQ + l] = __expf(t[ch][l] - m) * inv;
        __syncthreads();
    }
}

// ---------------------------------------------------------------------------
// E[b,i,j] = exp(-(i-j)^2 / (1e-5 + 2*std_i^2)) * w[b,j]   (row-max folded out)
// ---------------------------------------------------------------------------
__global__ __launch_bounds__(256) void build_e_kernel(const float* __restrict__ sarr,
                                                      const float* __restrict__ warr,
                                                      _Float16* __restrict__ E) {
    const int i = blockIdx.x, b = blockIdx.y;
    const float std = 0.1f * (float)LSEQ * sarr[b * LSEQ + i];
    const float inv = 1.0f / (1e-5f + 2.0f * std * std);
    _Float16* row = E + ((size_t)b * LSEQ + i) * LSEQ;
    const float* wr = warr + (size_t)b * LSEQ;
    for (int j = threadIdx.x; j < LSEQ; j += 256) {
        float d = (float)(i - j);
        row[j] = (_Float16)(__expf(-d * d * inv) * wr[j]);
    }
}

// ---------------------------------------------------------------------------
// out[b,i,d] = sum_j E[b,i,j] * feats16[b,j,d]  (WMMA GEMM, f32 out)
// ---------------------------------------------------------------------------
__global__ __launch_bounds__(128) void att_gemm_kernel(
    const _Float16* __restrict__ E,   // [B][1024][1024]
    const _Float16* __restrict__ F,   // [B][1024][768]
    float* __restrict__ out)          // [B][1024][768]
{
    __shared__ __align__(16) _Float16 As[64][LDT];
    __shared__ __align__(16) _Float16 Bt[64][LDT];    // TRANSPOSED: [col][k]
    const int d0 = blockIdx.x * 64;
    const int i0 = blockIdx.y * 64;
    const int b  = blockIdx.z;
    const int tid  = threadIdx.x;
    const int wave = tid >> 5;
    const int lane = tid & 31;
    const int mrow = (lane >> 4) << 3;

    v8f acc[4];
#pragma unroll
    for (int j = 0; j < 4; ++j)
#pragma unroll
        for (int r = 0; r < 8; ++r) acc[j][r] = 0.f;

    for (int kc = 0; kc < LSEQ / 32; ++kc) {
        {
            const _Float16* src = E + ((size_t)b * LSEQ + i0) * LSEQ + ((size_t)kc << 5);
            for (int idx = tid; idx < 64 * 4; idx += 128) {
                int r = idx >> 2, cc = idx & 3;
                *(uint4*)&As[r][cc * 8] = *(const uint4*)(src + (size_t)r * LSEQ + cc * 8);
            }
        }
        {
            const _Float16* src = F + ((size_t)b * LSEQ + ((size_t)kc << 5)) * DIN + d0;
            for (int idx = tid; idx < 32 * 8; idx += 128) {
                int r = idx >> 3, cc = idx & 7;
                uint4 q = *(const uint4*)(src + (size_t)r * DIN + cc * 8);
                const _Float16* h = (const _Float16*)&q;
#pragma unroll
                for (int i = 0; i < 8; ++i) Bt[cc * 8 + i][r] = h[i];
            }
        }
        // prefetch next K-chunk
        if (kc + 1 < LSEQ / 32) {
            if (tid < 32) {
                __builtin_prefetch(F + ((size_t)b * LSEQ + (((size_t)kc + 1) << 5) + tid) * DIN + d0, 0, 3);
            } else if (tid < 96) {
                int r = tid - 32;   // 0..63 -> 64 A rows
                __builtin_prefetch(E + ((size_t)b * LSEQ + i0 + r) * LSEQ + (((size_t)kc + 1) << 5), 0, 3);
            }
        }
        __syncthreads();
        v16h a = load_a_frag(&As[wave * 16][0], LDT);
#pragma unroll
        for (int j = 0; j < 4; ++j) {
            v16h bf = load_bt_frag(&Bt[0][0], j * 16);
            acc[j] = __builtin_amdgcn_wmma_f32_16x16x32_f16(
                false, a, false, bf, (short)0, acc[j], false, false);
        }
        __syncthreads();
    }

    float* dst = out + ((size_t)b * LSEQ + i0 + wave * 16 + mrow) * DIN + d0 + (lane & 15);
#pragma unroll
    for (int j = 0; j < 4; ++j)
#pragma unroll
        for (int r = 0; r < 8; ++r)
            dst[(size_t)r * DIN + j * 16] = acc[j][r];
}

// ---------------------------------------------------------------------------
extern "C" void kernel_launch(void* const* d_in, const int* in_sizes, int n_in,
                              void* d_out, int out_size, void* d_ws, size_t ws_size,
                              hipStream_t stream) {
    const float* feats   = (const float*)d_in[0];
    const float* conv1_w = (const float*)d_in[1];
    const float* conv1_b = (const float*)d_in[2];
    const float* ln1_w   = (const float*)d_in[3];
    const float* ln1_b   = (const float*)d_in[4];
    const float* conv2_w = (const float*)d_in[5];
    const float* conv2_b = (const float*)d_in[6];
    const float* ln2_w   = (const float*)d_in[7];
    const float* ln2_b   = (const float*)d_in[8];
    const float* conv3_w = (const float*)d_in[9];
    const float* conv3_b = (const float*)d_in[10];
    const float* ln3_w   = (const float*)d_in[11];
    const float* ln3_b   = (const float*)d_in[12];

    char* ws = (char*)d_ws;
    size_t off = 0;
    auto alloc = [&](size_t bytes) -> void* {
        void* p = ws + off;
        off += (bytes + 255) & ~(size_t)255;
        return p;
    };
    _Float16* w1h  = (_Float16*)alloc((size_t)3 * D2 * DIN * 2);
    _Float16* w2h  = (_Float16*)alloc((size_t)3 * D2 * D2 * 2);
    _Float16* xp0  = (_Float16*)alloc((size_t)NB * DIN * LPAD * 2);
    _Float16* xp1  = (_Float16*)alloc((size_t)NB * D2 * LPAD * 2);   // reused as E later
    _Float16* xp2  = (_Float16*)alloc((size_t)NB * D2 * LPAD * 2);
    _Float16* f16  = (_Float16*)alloc((size_t)NB * LSEQ * DIN * 2);
    float*    stats = (float*)alloc(64 * sizeof(float));             // [layer][b][sum,sumsq]
    float*    sbuf  = (float*)alloc((size_t)NB * LSEQ * sizeof(float));
    float*    wbuf  = (float*)alloc((size_t)NB * LSEQ * sizeof(float));
    _Float16* Eh = xp1;  // Xpad1 is dead after conv2 consumes it

    init_kernel<<<1920, 256, 0, stream>>>(stats, xp0, xp1, xp2);
    pack_w_kernel<<<4096, 256, 0, stream>>>(conv1_w, w1h, D2, DIN);
    pack_w_kernel<<<4096, 256, 0, stream>>>(conv2_w, w2h, D2, D2);
    pack_feats_kernel<<<8192, 256, 0, stream>>>(feats, xp0, f16);

    conv_gemm_kernel<<<dim3(LSEQ / 64, D2 / 64, NB), 128, 0, stream>>>(w1h, xp0, conv1_b, xp1, D2, DIN);
    ln_stats_kernel<<<dim3(96, NB), 256, 0, stream>>>(xp1, stats, D2);
    ln_apply_kernel<<<dim3(512, NB), 256, 0, stream>>>(xp1, stats, ln1_w, ln1_b, D2);

    conv_gemm_kernel<<<dim3(LSEQ / 64, D2 / 64, NB), 128, 0, stream>>>(w2h, xp1, conv2_b, xp2, D2, D2);
    ln_stats_kernel<<<dim3(96, NB), 256, 0, stream>>>(xp2, stats + 32, D2);
    ln_apply_kernel<<<dim3(512, NB), 256, 0, stream>>>(xp2, stats + 32, ln2_w, ln2_b, D2);

    head_kernel<<<NB, 256, 0, stream>>>(xp2, conv3_w, conv3_b, ln3_w, ln3_b, sbuf, wbuf);
    build_e_kernel<<<dim3(LSEQ, NB), 256, 0, stream>>>(sbuf, wbuf, Eh);
    att_gemm_kernel<<<dim3(DIN / 64, LSEQ / 64, NB), 128, 0, stream>>>(Eh, f16, (float*)d_out);
}
